// HybridGNNTransformer_31233002177229
// MI455X (gfx1250) — matrix operand order repaired and checked by reading, not compiled, not run on hardware
//
#include <hip/hip_runtime.h>
#include <hip/hip_bf16.h>

#define N_NODES 4096
#define IN_CH   128
#define H_CH    256
#define NC_CLS  16
#define NH_H    4
#define HD_D    64

typedef __attribute__((ext_vector_type(16))) _Float16 v16h;
typedef __attribute__((ext_vector_type(8)))  _Float16 v8h;
typedef __attribute__((ext_vector_type(2)))  _Float16 v2h;
typedef __attribute__((ext_vector_type(8)))  float    v8f;

// ---------------------------------------------------------------------------
// helpers
// ---------------------------------------------------------------------------
__device__ __forceinline__ v16h pack16(float4 a, float4 b, float4 c, float4 d) {
  v16h r;
  r[0] = (_Float16)a.x; r[1] = (_Float16)a.y; r[2]  = (_Float16)a.z; r[3]  = (_Float16)a.w;
  r[4] = (_Float16)b.x; r[5] = (_Float16)b.y; r[6]  = (_Float16)b.z; r[7]  = (_Float16)b.w;
  r[8] = (_Float16)c.x; r[9] = (_Float16)c.y; r[10] = (_Float16)c.z; r[11] = (_Float16)c.w;
  r[12] = (_Float16)d.x; r[13] = (_Float16)d.y; r[14] = (_Float16)d.z; r[15] = (_Float16)d.w;
  return r;
}

// A fragment 16x32: lane&15 = row; lane>>4 selects K halves
// {0..15} vs {16..31}. Requires 16B-aligned rows, K%32==0, M%16==0.
__device__ __forceinline__ v16h load_a_frag(const float* __restrict__ A, int lda,
                                            int row0, int k0, int lane) {
  const float* p = A + (size_t)(row0 + (lane & 15)) * lda + k0 + ((lane >> 4) << 4);
  float4 a0 = *(const float4*)(p);
  float4 a1 = *(const float4*)(p + 4);
  float4 a2 = *(const float4*)(p + 8);
  float4 a3 = *(const float4*)(p + 12);
  return pack16(a0, a1, a2, a3);
}

// asynchronous global -> LDS copy of 16 bytes per lane (ASYNCcnt-tracked)
__device__ __forceinline__ void async_copy_b128(void* lds_ptr, const void* g) {
  unsigned lds_off = (unsigned)(uintptr_t)lds_ptr;  // low 32 bits = LDS address
  unsigned long long ga = (unsigned long long)(uintptr_t)g;
  asm volatile("global_load_async_to_lds_b128 %0, %1, off"
               :: "v"(lds_off), "v"(ga) : "memory");
}

__device__ __forceinline__ void wait_async_all() {
  asm volatile("s_wait_asynccnt 0x0" ::: "memory");
}

// 16-lane butterfly reductions in pure VALU via DPP16 (rows of 16 lanes ==
// the WMMA C-layout column groups). Sequence: xor1, xor2, xor7(within 8),
// xor15(within 16) -- valid for commutative reduction.
// DPP control must be a constant -> template parameter.
template <int CTRL>
__device__ __forceinline__ float dpp_perm_f(float v) {
  return __int_as_float(__builtin_amdgcn_update_dpp(
      0, __float_as_int(v), CTRL, 0xf, 0xf, true));
}
__device__ __forceinline__ float redmax16(float v) {
  v = fmaxf(v, dpp_perm_f<0xB1>(v));   // quad_perm [1,0,3,2]  (xor 1)
  v = fmaxf(v, dpp_perm_f<0x4E>(v));   // quad_perm [2,3,0,1]  (xor 2)
  v = fmaxf(v, dpp_perm_f<0x141>(v));  // row_half_mirror      (xor 7)
  v = fmaxf(v, dpp_perm_f<0x140>(v));  // row_mirror           (xor 15)
  return v;
}
__device__ __forceinline__ float redsum16(float v) {
  v += dpp_perm_f<0xB1>(v);
  v += dpp_perm_f<0x4E>(v);
  v += dpp_perm_f<0x141>(v);
  v += dpp_perm_f<0x140>(v);
  return v;
}

// ---------------------------------------------------------------------------
// one-time weight packing: f32 weights -> f16, layout [Nout, K] so each lane's
// 16 contraction elements are one contiguous 32B v16h load.
// trans==1: input is [K,Nout] row-major; trans==0: input already [Nout,K].
// ---------------------------------------------------------------------------
__global__ void pack_w_kernel(const float* __restrict__ W, _Float16* __restrict__ Wp,
                              int K, int Nn, int trans) {
  int t = blockIdx.x * blockDim.x + threadIdx.x;
  if (t >= K * Nn) return;
  int n = t / K, k = t - n * K;
  float v = trans ? W[(size_t)k * Nn + n] : W[(size_t)n * K + k];
  Wp[t] = (_Float16)v;
}

// ---------------------------------------------------------------------------
// WMMA GEMM: C[M,Nn] = act(A[M,K] @ Bp^T + bias), Bp is f16 [Nn,K] packed.
// One wave computes a 16x64 tile. M%16==0, K%32==0 assumed.
// ACT: 0 none, 1 relu, 2 sigmoid. COLCHK: bounds-check columns vs Nn.
// ---------------------------------------------------------------------------
template <int ACT, int COLCHK, typename OutT>
__global__ void __launch_bounds__(256)
gemm_wmma_kernel(const float* __restrict__ A, const _Float16* __restrict__ Bp,
                 const float* __restrict__ bias, OutT* __restrict__ C,
                 int M, int K, int Nn) {
  const int lane = threadIdx.x & 31;
  const int wave = blockIdx.x * 8 + (threadIdx.x >> 5);
  const int ntile64 = (Nn + 63) >> 6;
  const int row0 = (wave / ntile64) << 4;
  const int n0b  = (wave % ntile64) << 6;
  if (row0 >= M) return;

  const int cc = lane & 15;
  v8f acc[4] = {v8f{}, v8f{}, v8f{}, v8f{}};

  for (int k0 = 0; k0 < K; k0 += 32) {
    v16h a = load_a_frag(A, K, row0, k0, lane);
    const int kb = k0 + ((lane >> 4) << 4);   // 16 contiguous K per lane half
#pragma unroll
    for (int t = 0; t < 4; ++t) {
      const int n0 = n0b + (t << 4);
      if (n0 < Nn) {
        const int col = n0 + cc;
        v16h b;
        if (COLCHK && col >= Nn) b = v16h{};
        else b = *(const v16h*)(Bp + (size_t)col * K + kb);   // 32B contiguous
        acc[t] = __builtin_amdgcn_wmma_f32_16x16x32_f16(
            false, a, false, b, (short)0, acc[t], false, false);
      }
    }
  }

  const int rb = (lane >> 4) << 3;
#pragma unroll
  for (int t = 0; t < 4; ++t) {
    const int col = n0b + (t << 4) + cc;
    if ((t << 4) + n0b < Nn && (!COLCHK || col < Nn)) {
      float bv = bias ? bias[col] : 0.f;
#pragma unroll
      for (int r = 0; r < 8; ++r) {
        float v = acc[t][r] + bv;
        if (ACT == 1) v = fmaxf(v, 0.f);
        else if (ACT == 2) v = 1.f / (1.f + __expf(-v));
        C[(size_t)(row0 + rb + r) * Nn + col] = (OutT)v;
      }
    }
  }
}

// ---------------------------------------------------------------------------
// Flash attention over f16 qkv [N, 768]. Block = 4 waves = (head, 64 q rows);
// each wave owns 16 q rows. K tiles async-copied (f16, b128) into LDS; V tiles
// staged transposed (paired b32 stores) so P@V B-fragments are contiguous
// v16h LDS loads. Softmax scale folded into f32 scores; row reductions in DPP.
// ---------------------------------------------------------------------------
__global__ void __launch_bounds__(128)
flash_attn_kernel(const _Float16* __restrict__ qkv, float* __restrict__ ctx, int Nn) {
  const int lane = threadIdx.x & 31;
  const int w    = threadIdx.x >> 5;
  const int qblocks = Nn >> 6;
  const int h  = blockIdx.x / qblocks;
  const int qb = blockIdx.x % qblocks;
  const int q0 = (qb << 6) + (w << 4);

  __shared__ _Float16 Ksh[32][HD_D];      // 4 KB, [key][d]
  __shared__ _Float16 Vt[HD_D][32];       // 4 KB, transposed [d][key]
  __shared__ _Float16 Psh[4][16][32];     // 4 KB, per-wave P staging

  const int   ldq   = 3 * H_CH;           // 768
  const float scale = 0.125f;             // 1/sqrt(64), applied to scores

  // Q fragments (A-layout), pure f16 copies
  v16h qa[2];
  {
    const _Float16* qp0 = qkv + (size_t)(q0 + (lane & 15)) * ldq + h * HD_D
                          + ((lane >> 4) << 4);
#pragma unroll
    for (int t = 0; t < 2; ++t) {
      v8h lo = *(const v8h*)(qp0 + t * 32);
      v8h hi = *(const v8h*)(qp0 + t * 32 + 8);
#pragma unroll
      for (int j = 0; j < 8; ++j) { qa[t][j] = lo[j]; qa[t][8 + j] = hi[j]; }
    }
  }

  v8f o[4] = {v8f{}, v8f{}, v8f{}, v8f{}};
  float mrow[8], lrow[8];
#pragma unroll
  for (int r = 0; r < 8; ++r) { mrow[r] = -1e30f; lrow[r] = 0.f; }

  const int cc = lane & 15;
  const int rb = (lane >> 4) << 3;

  for (int kt = 0; kt < Nn; kt += 32) {
    // K tile: async f16 b128 copies
#pragma unroll
    for (int it = 0; it < 2; ++it) {
      int linear = (it * 128 + threadIdx.x) << 3;   // half index within tile
      int j = linear >> 6, d = linear & 63;
      async_copy_b128(&Ksh[j][d],
                      qkv + H_CH + (size_t)(kt + j) * ldq + h * HD_D + d);
    }
    // V tile transposed: 2 adjacent keys x 8 d per thread -> 8 paired b32 stores
    {
      int jp = threadIdx.x >> 3;              // 0..15 -> keys 2jp, 2jp+1
      int d0 = (threadIdx.x & 7) << 3;        // 0,8,...,56
      const _Float16* vbase = qkv + 2 * H_CH + h * HD_D + d0;
      v8h va = *(const v8h*)(vbase + (size_t)(kt + 2 * jp) * ldq);
      v8h vb = *(const v8h*)(vbase + (size_t)(kt + 2 * jp + 1) * ldq);
#pragma unroll
      for (int i = 0; i < 8; ++i) {
        v2h pr; pr[0] = va[i]; pr[1] = vb[i];
        *(v2h*)&Vt[d0 + i][2 * jp] = pr;
      }
    }
    if (kt + 32 < Nn) {
      size_t nb = (size_t)(kt + 32 + lane) * ldq + h * HD_D;
      __builtin_prefetch(qkv + H_CH + nb, 0, 0);
      __builtin_prefetch(qkv + 2 * H_CH + nb, 0, 0);
    }
    wait_async_all();
    __syncthreads();

    // scores S = Q @ K^T : two 16x16 tiles, 2 K-steps each; B-frag = v16h load
    v8f s[2] = {v8f{}, v8f{}};
#pragma unroll
    for (int jt = 0; jt < 2; ++jt) {
      const int col = (jt << 4) + cc;
#pragma unroll
      for (int dt = 0; dt < 2; ++dt) {
        v16h bk = *(const v16h*)&Ksh[col][(dt << 5) + ((lane >> 4) << 4)];
        s[jt] = __builtin_amdgcn_wmma_f32_16x16x32_f16(
            false, qa[dt], false, bk, (short)0, s[jt], false, false);
      }
    }

    // online softmax; per-lane stats index rows rb+r (matches C layout);
    // 16-lane row reductions via DPP butterflies (VALU, no LDS traffic)
#pragma unroll
    for (int r = 0; r < 8; ++r) {
      float v0 = s[0][r] * scale, v1 = s[1][r] * scale;
      float mx = redmax16(fmaxf(v0, v1));
      float mnew = fmaxf(mrow[r], mx);
      float p0 = __expf(v0 - mnew), p1 = __expf(v1 - mnew);
      float sum = redsum16(p0 + p1);
      float alpha = __expf(mrow[r] - mnew);
      mrow[r] = mnew;
      lrow[r] = lrow[r] * alpha + sum;
#pragma unroll
      for (int t = 0; t < 4; ++t) o[t][r] *= alpha;
      Psh[w][rb + r][cc]      = (_Float16)p0;   // C-layout -> LDS
      Psh[w][rb + r][16 + cc] = (_Float16)p1;
    }

    // reload P as A-layout fragment (same-wave LDS is in-order)
    v16h pa;
    {
      const _Float16* pp = &Psh[w][lane & 15][(lane >> 4) << 4];
      v8h lo = *(const v8h*)(pp);
      v8h hi = *(const v8h*)(pp + 8);
#pragma unroll
      for (int j = 0; j < 8; ++j) { pa[j] = lo[j]; pa[8 + j] = hi[j]; }
    }

    // O += P @ V : one K-step (32 keys), 4 d-chunks; B-frag = v16h from Vt
#pragma unroll
    for (int t = 0; t < 4; ++t) {
      v16h bv = *(const v16h*)&Vt[(t << 4) + cc][(lane >> 4) << 4];
      o[t] = __builtin_amdgcn_wmma_f32_16x16x32_f16(
          false, pa, false, bv, (short)0, o[t], false, false);
    }
    __syncthreads();
  }

  // normalize and write context
#pragma unroll
  for (int t = 0; t < 4; ++t) {
#pragma unroll
    for (int r = 0; r < 8; ++r) {
      ctx[(size_t)(q0 + rb + r) * H_CH + h * HD_D + (t << 4) + cc] =
          o[t][r] / lrow[r];
    }
  }
}

// ---------------------------------------------------------------------------
// GCN aggregation + misc elementwise kernels
// ---------------------------------------------------------------------------
__global__ void deg_init_kernel(float* __restrict__ deg, int n) {
  int i = blockIdx.x * blockDim.x + threadIdx.x;
  if (i < n) deg[i] = 1.0f;  // self-loop weight
}

__global__ void deg_accum_kernel(const int* __restrict__ ei,
                                 const float* __restrict__ ew,
                                 float* __restrict__ deg, int E) {
  int e = blockIdx.x * blockDim.x + threadIdx.x;
  if (e < E) atomicAdd(&deg[ei[E + e]], ew[e]);
}

__global__ void dinv_kernel(const float* __restrict__ deg,
                            float* __restrict__ dinv, int n) {
  int i = blockIdx.x * blockDim.x + threadIdx.x;
  if (i < n) { float d = deg[i]; dinv[i] = (d > 0.f) ? rsqrtf(d) : 0.f; }
}

__global__ void selfloop_init_kernel(const float* __restrict__ dinv,
                                     const float* __restrict__ h,
                                     float* __restrict__ out, int n) {
  int t = blockIdx.x * blockDim.x + threadIdx.x;
  if (t < n * H_CH) {
    float dv = dinv[t >> 8];
    out[t] = dv * dv * h[t];
  }
}

__global__ void edge_scatter_kernel(const int* __restrict__ ei,
                                    const float* __restrict__ ew,
                                    const float* __restrict__ dinv,
                                    const float* __restrict__ h,
                                    float* __restrict__ out, int E) {
  long long t = (long long)blockIdx.x * blockDim.x + threadIdx.x;
  const int per = H_CH / 4;
  int e  = (int)(t / per);
  int c4 = (int)(t % per) * 4;
  if (e >= E) return;
  int s = ei[e], d = ei[E + e];
  float nrm = dinv[s] * ew[e] * dinv[d];
  const float4 hv = *(const float4*)(h + (size_t)s * H_CH + c4);
  float* op = out + (size_t)d * H_CH + c4;
  atomicAdd(op + 0, nrm * hv.x);
  atomicAdd(op + 1, nrm * hv.y);
  atomicAdd(op + 2, nrm * hv.z);
  atomicAdd(op + 3, nrm * hv.w);
}

__global__ void bias_act_kernel(float* __restrict__ x, const float* __restrict__ b,
                                int n, int act) {
  int t = blockIdx.x * blockDim.x + threadIdx.x;
  if (t < n * H_CH) {
    float v = x[t] + b[t & (H_CH - 1)];
    if (act == 1) v = fmaxf(v, 0.f);
    x[t] = v;
  }
}

__global__ void add_kernel(const float* __restrict__ a, const float* __restrict__ b,
                           float* __restrict__ c, int n) {
  int t = blockIdx.x * blockDim.x + threadIdx.x;
  if (t < n) c[t] = a[t] + b[t];
}

__global__ void __launch_bounds__(256)
layernorm_add_kernel(const float* __restrict__ xa, const float* __restrict__ xb,
                     const float* __restrict__ g, const float* __restrict__ bta,
                     float* __restrict__ out) {
  __shared__ float red[256];
  const int row = blockIdx.x, c = threadIdx.x;
  const size_t idx = (size_t)row * H_CH + c;
  float v = xa[idx] + xb[idx];
  red[c] = v; __syncthreads();
  for (int s = 128; s > 0; s >>= 1) { if (c < s) red[c] += red[c + s]; __syncthreads(); }
  float mean = red[0] * (1.f / H_CH);
  __syncthreads();
  float d = v - mean;
  red[c] = d * d; __syncthreads();
  for (int s = 128; s > 0; s >>= 1) { if (c < s) red[c] += red[c + s]; __syncthreads(); }
  float var = red[0] * (1.f / H_CH);
  out[idx] = d * rsqrtf(var + 1e-5f) * g[c] + bta[c];
}

__global__ void log_softmax16_kernel(float* __restrict__ out, int n) {
  int wid  = (blockIdx.x * blockDim.x + threadIdx.x) >> 5;
  int lane = threadIdx.x & 31;
  if (wid >= n) return;
  float v = (lane < NC_CLS) ? out[wid * NC_CLS + lane] : -1e30f;
  float mx = redmax16(v);                         // lanes 0..15 reduce together
  float e = (lane < NC_CLS) ? __expf(v - mx) : 0.f;
  float sum = redsum16(e);
  if (lane < NC_CLS) out[wid * NC_CLS + lane] = v - mx - __logf(sum);
}

// ---------------------------------------------------------------------------
// host side
// ---------------------------------------------------------------------------
static inline int gemm_blocks(int M, int Nn) {
  return (((M >> 4) * ((Nn + 63) >> 6)) + 7) >> 3;  // 8 waves per block
}

extern "C" void kernel_launch(void* const* d_in, const int* in_sizes, int n_in,
                              void* d_out, int out_size, void* d_ws, size_t ws_size,
                              hipStream_t stream) {
  const float* x        = (const float*)d_in[0];
  const int*   ei       = (const int*)  d_in[1];
  const float* ea       = (const float*)d_in[2];
  const float* W_gcn1   = (const float*)d_in[3];
  const float* b_gcn1   = (const float*)d_in[4];
  const float* W_gcn2   = (const float*)d_in[5];
  const float* b_gcn2   = (const float*)d_in[6];
  const float* in_proj_w  = (const float*)d_in[7];
  const float* in_proj_b  = (const float*)d_in[8];
  const float* out_proj_w = (const float*)d_in[9];
  const float* out_proj_b = (const float*)d_in[10];
  const float* ln1_g    = (const float*)d_in[11];
  const float* ln1_b    = (const float*)d_in[12];
  const float* ffn_w1   = (const float*)d_in[13];
  const float* ffn_b1   = (const float*)d_in[14];
  const float* ffn_w2   = (const float*)d_in[15];
  const float* ffn_b2   = (const float*)d_in[16];
  const float* ln2_g    = (const float*)d_in[17];
  const float* ln2_b    = (const float*)d_in[18];
  const float* W_cls    = (const float*)d_in[19];
  const float* b_cls    = (const float*)d_in[20];
  const float* W_d1     = (const float*)d_in[21];
  const float* b_d1     = (const float*)d_in[22];
  const float* W_d2     = (const float*)d_in[23];
  const float* b_d2     = (const float*)d_in[24];

  const int Nn = N_NODES;
  const int E  = in_sizes[1] / 2;

  // ---- workspace carve-up (f32 region, then f16 packed weights) ----
  float* ws = (float*)d_ws;
  size_t o = 0;
  float* deg  = ws + o; o += Nn;
  float* dinv = ws + o; o += Nn;
  float* bufA = ws + o; o += (size_t)Nn * H_CH;
  float* bufB = ws + o; o += (size_t)Nn * H_CH;
  float* xl   = ws + o; o += (size_t)Nn * H_CH;            // x_local
  float* qd   = ws + o; o += (size_t)Nn * 2 * H_CH;        // qkv(f16) / d1(f32)
  float* ctxb = ws + o; o += (size_t)Nn * H_CH;
  float* x1   = ws + o; o += (size_t)Nn * H_CH;            // reused as x_final
  float* ffn1 = ws + o; o += (size_t)Nn * 4 * H_CH;

  _Float16* hbase = (_Float16*)(ws + o);
  size_t ho = 0;
  _Float16* p_gcn1 = hbase + ho; ho += (size_t)H_CH * IN_CH;
  _Float16* p_gcn2 = hbase + ho; ho += (size_t)H_CH * H_CH;
  _Float16* p_inp  = hbase + ho; ho += (size_t)3 * H_CH * H_CH;
  _Float16* p_outp = hbase + ho; ho += (size_t)H_CH * H_CH;
  _Float16* p_f1   = hbase + ho; ho += (size_t)4 * H_CH * H_CH;
  _Float16* p_f2   = hbase + ho; ho += (size_t)H_CH * 4 * H_CH;
  _Float16* p_cls  = hbase + ho; ho += (size_t)NC_CLS * H_CH;
  _Float16* p_d1   = hbase + ho; ho += (size_t)2 * H_CH * H_CH;
  _Float16* p_d2   = hbase + ho; ho += (size_t)3 * 2 * H_CH;

  _Float16* qkvh = (_Float16*)qd;                          // f16 [N, 768]
  float*    d1f  = qd;                                     // f32 [N, 512]

  float* out_cls  = (float*)d_out;                          // [N,16]
  float* out_reco = (float*)d_out + (size_t)Nn * NC_CLS;    // [N,3]

  const int elemNH = Nn * H_CH;
  const int eb = 256;
#define GRID1(n) (((n) + eb - 1) / eb)

  // ---- pack all weights to f16 [Nout,K] (one-time per launch) ----
  pack_w_kernel<<<GRID1(IN_CH * H_CH), eb, 0, stream>>>(W_gcn1, p_gcn1, IN_CH, H_CH, 1);
  pack_w_kernel<<<GRID1(H_CH * H_CH), eb, 0, stream>>>(W_gcn2, p_gcn2, H_CH, H_CH, 1);
  pack_w_kernel<<<GRID1(3 * H_CH * H_CH), eb, 0, stream>>>(in_proj_w, p_inp, H_CH, 3 * H_CH, 0);
  pack_w_kernel<<<GRID1(H_CH * H_CH), eb, 0, stream>>>(out_proj_w, p_outp, H_CH, H_CH, 0);
  pack_w_kernel<<<GRID1(4 * H_CH * H_CH), eb, 0, stream>>>(ffn_w1, p_f1, H_CH, 4 * H_CH, 1);
  pack_w_kernel<<<GRID1(4 * H_CH * H_CH), eb, 0, stream>>>(ffn_w2, p_f2, 4 * H_CH, H_CH, 1);
  pack_w_kernel<<<GRID1(H_CH * NC_CLS), eb, 0, stream>>>(W_cls, p_cls, H_CH, NC_CLS, 1);
  pack_w_kernel<<<GRID1(2 * H_CH * H_CH), eb, 0, stream>>>(W_d1, p_d1, H_CH, 2 * H_CH, 1);
  pack_w_kernel<<<GRID1(2 * H_CH * 3), eb, 0, stream>>>(W_d2, p_d2, 2 * H_CH, 3, 1);

  // ---- degree / normalization ----
  deg_init_kernel<<<GRID1(Nn), eb, 0, stream>>>(deg, Nn);
  deg_accum_kernel<<<GRID1(E), eb, 0, stream>>>(ei, ea, deg, E);
  dinv_kernel<<<GRID1(Nn), eb, 0, stream>>>(deg, dinv, Nn);

  const long long scat_threads = (long long)E * (H_CH / 4);
  const int scat_blocks = (int)((scat_threads + eb - 1) / eb);

  // ---- GCN layer 1: relu(aggregate(x @ W1) + b1) ----
  gemm_wmma_kernel<0, 0, float><<<gemm_blocks(Nn, H_CH), 256, 0, stream>>>(
      x, p_gcn1, nullptr, bufA, Nn, IN_CH, H_CH);
  selfloop_init_kernel<<<GRID1(elemNH), eb, 0, stream>>>(dinv, bufA, bufB, Nn);
  edge_scatter_kernel<<<scat_blocks, eb, 0, stream>>>(ei, ea, dinv, bufA, bufB, E);
  bias_act_kernel<<<GRID1(elemNH), eb, 0, stream>>>(bufB, b_gcn1, Nn, 1);

  // ---- GCN layer 2: aggregate(xl1 @ W2) + b2 -> x_local ----
  gemm_wmma_kernel<0, 0, float><<<gemm_blocks(Nn, H_CH), 256, 0, stream>>>(
      bufB, p_gcn2, nullptr, bufA, Nn, H_CH, H_CH);
  selfloop_init_kernel<<<GRID1(elemNH), eb, 0, stream>>>(dinv, bufA, xl, Nn);
  edge_scatter_kernel<<<scat_blocks, eb, 0, stream>>>(ei, ea, dinv, bufA, xl, E);
  bias_act_kernel<<<GRID1(elemNH), eb, 0, stream>>>(xl, b_gcn2, Nn, 0);

  // ---- QKV projection -> f16 qkv ----
  gemm_wmma_kernel<0, 0, _Float16><<<gemm_blocks(Nn, 3 * H_CH), 256, 0, stream>>>(
      xl, p_inp, in_proj_b, qkvh, Nn, H_CH, 3 * H_CH);

  // ---- flash attention ----
  flash_attn_kernel<<<NH_H * (Nn >> 6), 128, 0, stream>>>(qkvh, ctxb, Nn);

  // ---- out_proj, residual + LN1 ----
  gemm_wmma_kernel<0, 0, float><<<gemm_blocks(Nn, H_CH), 256, 0, stream>>>(
      ctxb, p_outp, out_proj_b, bufA, Nn, H_CH, H_CH);
  layernorm_add_kernel<<<Nn, 256, 0, stream>>>(xl, bufA, ln1_g, ln1_b, x1);

  // ---- FFN ----
  gemm_wmma_kernel<1, 0, float><<<gemm_blocks(Nn, 4 * H_CH), 256, 0, stream>>>(
      x1, p_f1, ffn_b1, ffn1, Nn, H_CH, 4 * H_CH);
  gemm_wmma_kernel<0, 0, float><<<gemm_blocks(Nn, H_CH), 256, 0, stream>>>(
      ffn1, p_f2, ffn_b2, bufA, Nn, 4 * H_CH, H_CH);
  layernorm_add_kernel<<<Nn, 256, 0, stream>>>(x1, bufA, ln2_g, ln2_b, bufB);

  // ---- x_final = x_local + x_global ----
  add_kernel<<<GRID1(elemNH), eb, 0, stream>>>(xl, bufB, x1, elemNH);

  // ---- classifier head ----
  gemm_wmma_kernel<0, 0, float><<<gemm_blocks(Nn, NC_CLS), 256, 0, stream>>>(
      x1, p_cls, b_cls, out_cls, Nn, H_CH, NC_CLS);
  log_softmax16_kernel<<<GRID1(Nn * 32), eb, 0, stream>>>(out_cls, Nn);

  // ---- semantic decoder (d1 reuses qkv region as f32) ----
  gemm_wmma_kernel<1, 0, float><<<gemm_blocks(Nn, 2 * H_CH), 256, 0, stream>>>(
      x1, p_d1, b_d1, d1f, Nn, H_CH, 2 * H_CH);
  gemm_wmma_kernel<2, 1, float><<<gemm_blocks(Nn, 3), 256, 0, stream>>>(
      d1f, p_d2, b_d2, out_reco, Nn, 2 * H_CH, 3);
}
